// LinearRNNEncDec_53128745451769
// MI455X (gfx1250) — matrix operand order repaired and checked by reading, not compiled, not run on hardware
//
#include <hip/hip_runtime.h>
#include <stdint.h>

// MI455X (gfx1250) linear RNN enc/dec via chunked-scan bf16 WMMA.
// All GEMMs: D = A @ Bw^T (+ addend), Bw in nn.Linear [N,K] layout,
// V_WMMA_F32_16X16X32_BF16, fp32 accumulate, wave32.
// Each wave computes a 16x64 tile: 1 A fragment reused by 4 independent WMMAs.

typedef __bf16 bf16_t;
typedef __attribute__((ext_vector_type(16))) __bf16 v16bf;
typedef __attribute__((ext_vector_type(8)))  float  v8f;
typedef __attribute__((ext_vector_type(8)))  __bf16 bf16x8;
typedef __attribute__((ext_vector_type(4)))  float  f32x4;

struct RowMap {
    long long base;   // element offset added to every row
    long long p0;     // multiplies (i >> shift)
    long long p1;     // multiplies (i & ((1<<shift)-1))
    int       shift;
};

__device__ __forceinline__ long long rowOff(const RowMap m, int i) {
    return m.base + (long long)(i >> m.shift) * m.p0 +
           (long long)(i & ((1 << m.shift) - 1)) * m.p1;
}

template <bool A_F32, bool OUT_F32>
__global__ __launch_bounds__(256)
void gemm_wmma_bf16(const void* __restrict__ Aptr, RowMap mapA,
                    const bf16_t* __restrict__ Bw,          // [N,K] bf16
                    const bf16_t* __restrict__ Add, RowMap mapC,
                    void* __restrict__ Dptr, RowMap mapD,
                    int M, int N, int K)
{
    const int lane = threadIdx.x & 31;
    const int wave = threadIdx.x >> 5;
    const int ng64   = N >> 6;               // 64-wide N groups
    const int mtiles = (M + 15) >> 4;
    const int tile = blockIdx.x * 8 + wave;
    if (tile >= ng64 * mtiles) return;
    const int mt = tile / ng64;
    const int ng = tile - mt * ng64;

    // A fragment: lane<16 holds row M=lane, K {0..7,16..23}; lane>=16 K {8..15,24..31}.
    // Out-of-range rows are clamped (their D rows are simply never stored).
    int ar = (mt << 4) + (lane & 15);
    if (ar >= M) ar = M - 1;
    const int akoff = (lane >> 4) << 3;          // 0 or 8
    // B fragment: lane%16 = column n; lanes<16 K {0..7,16..23}, lanes>=16 K {8..15,24..31}
    const int bkoff = (lane >> 4) << 3;          // 0 or 8

    const float*  Af = (const float*)Aptr;
    const bf16_t* Ab = (const bf16_t*)Aptr;
    const long long aoff = rowOff(mapA, ar);
    const bf16_t* brow0 = Bw + (long long)((ng << 6) + (lane & 15)) * K + bkoff;
    const long long bsub = (long long)16 * K;    // stride between the 4 N-subtiles

    v8f acc[4] = {v8f{}, v8f{}, v8f{}, v8f{}};

    for (int k0 = 0; k0 < K; k0 += 32) {
        // ---- A fragment ----
        v16bf a;
        if (A_F32) {
            const float* ap = Af + aoff + k0 + akoff;
            f32x4 l0 = *(const f32x4*)(ap);
            f32x4 l1 = *(const f32x4*)(ap + 4);
            f32x4 h0 = *(const f32x4*)(ap + 16);
            f32x4 h1 = *(const f32x4*)(ap + 20);
#pragma unroll
            for (int i = 0; i < 4; ++i) {
                a[i]      = (bf16_t)l0[i];
                a[4 + i]  = (bf16_t)l1[i];
                a[8 + i]  = (bf16_t)h0[i];
                a[12 + i] = (bf16_t)h1[i];
            }
        } else {
            const bf16_t* ap = Ab + aoff + k0 + akoff;
            bf16x8 lo = *(const bf16x8*)(ap);
            bf16x8 hi = *(const bf16x8*)(ap + 16);
#pragma unroll
            for (int i = 0; i < 8; ++i) { a[i] = lo[i]; a[8 + i] = hi[i]; }
        }
        // ---- 4 B fragments (independent N subtiles) ----
        v16bf b[4];
#pragma unroll
        for (int s = 0; s < 4; ++s) {
            const bf16_t* bp = brow0 + s * bsub + k0;
            bf16x8 lo = *(const bf16x8*)(bp);
            bf16x8 hi = *(const bf16x8*)(bp + 16);
#pragma unroll
            for (int i = 0; i < 8; ++i) { b[s][i] = lo[i]; b[s][8 + i] = hi[i]; }
        }
        // ---- prefetch next K chunk into near cache (global_prefetch_b8) ----
        if (k0 + 32 < K) {
            if (A_F32) __builtin_prefetch(Af + aoff + k0 + 32 + akoff, 0, 3);
            else       __builtin_prefetch(Ab + aoff + k0 + 32 + akoff, 0, 3);
#pragma unroll
            for (int s = 0; s < 4; ++s)
                __builtin_prefetch(brow0 + s * bsub + k0 + 32, 0, 3);
        }
        // ---- 4 independent WMMAs, no D->A/B hazards between them ----
#pragma unroll
        for (int s = 0; s < 4; ++s)
            acc[s] = __builtin_amdgcn_wmma_f32_16x16x32_bf16(
                false, a, false, b[s], (short)0, acc[s], false, false);
    }

    // ---- epilogue: D rows m = mt*16 + v + 8*(lane>=16), n per subtile ----
    const bool hasAdd = (Add != nullptr);
#pragma unroll
    for (int s = 0; s < 4; ++s) {
        const int n = (ng << 6) + (s << 4) + (lane & 15);
#pragma unroll
        for (int v = 0; v < 8; ++v) {
            const int m = (mt << 4) + v + ((lane >> 4) << 3);
            if (m >= M) continue;
            float val = acc[s][v];
            if (hasAdd) val += (float)Add[rowOff(mapC, m) + n];
            if (OUT_F32) ((float*)Dptr)[rowOff(mapD, m) + n] = val;
            else        ((bf16_t*)Dptr)[rowOff(mapD, m) + n] = (bf16_t)val;
        }
    }
}

__global__ __launch_bounds__(256)
void transpose_bf16(const bf16_t* __restrict__ in, bf16_t* __restrict__ out,
                    int R, int C)  // out[c][r] = in[r][c]; R,C multiples of 32
{
    __shared__ bf16_t tile[32][33];
    const int bx = blockIdx.x * 32;     // col base
    const int by = blockIdx.y * 32;     // row base
    const int tx = threadIdx.x & 31;
    const int ty = threadIdx.x >> 5;    // 0..7
    for (int r = ty; r < 32; r += 8)
        tile[r][tx] = in[(long long)(by + r) * C + bx + tx];
    __syncthreads();
    for (int r = ty; r < 32; r += 8)
        out[(long long)(bx + r) * R + by + tx] = tile[tx][r];
}

__global__ void f32_to_bf16_k(const float* __restrict__ in, bf16_t* __restrict__ out, int n) {
    int i = blockIdx.x * blockDim.x + threadIdx.x;
    if (i < n) out[i] = (bf16_t)in[i];
}

__global__ void zero_bf16_k(bf16_t* __restrict__ p, int n) {
    int i = blockIdx.x * blockDim.x + threadIdx.x;
    if (i < n) p[i] = (bf16_t)0.0f;
}

extern "C" void kernel_launch(void* const* d_in, const int* in_sizes, int n_in,
                              void* d_out, int out_size, void* d_ws, size_t ws_size,
                              hipStream_t stream)
{
    (void)in_sizes; (void)n_in; (void)out_size; (void)ws_size;
    constexpr int B = 8, L = 2048, DIN = 512, H = 1024, DOUT = 512, OUT_LEN = 512;
    constexpr int NC = 16, CENC = 128;   // encoder: 16 chunks x 128 steps
    constexpr int CDEC = 32;             // decoder: 16 chunks x 32 steps

    const float* x  = (const float*)d_in[0];
    const float* U  = (const float*)d_in[1];
    const float* W  = (const float*)d_in[2];
    const float* V  = (const float*)d_in[3];
    const float* Mw = (const float*)d_in[4];
    const float* cT = (const float*)d_in[5];
    float* out = (float*)d_out;

    char* ws = (char*)d_ws;
    size_t cur = 0;
    auto alloc = [&](size_t bytes) -> char* {
        cur = (cur + 255) & ~((size_t)255);
        char* p = ws + cur; cur += bytes; return p;
    };
    bf16_t* xu   = (bf16_t*)alloc((size_t)L * B * H * 2);        // row t*8+b
    bf16_t* outs = (bf16_t*)alloc((size_t)OUT_LEN * B * H * 2);  // row t*8+b
    bf16_t* Ub  = (bf16_t*)alloc((size_t)H * DIN * 2);
    bf16_t* Wb  = (bf16_t*)alloc((size_t)H * H * 2);
    bf16_t* Vb  = (bf16_t*)alloc((size_t)H * H * 2);
    bf16_t* Mb  = (bf16_t*)alloc((size_t)H * H * 2);
    bf16_t* cTb = (bf16_t*)alloc((size_t)DOUT * H * 2);
    bf16_t* Sa  = (bf16_t*)alloc((size_t)H * H * 2);
    bf16_t* Sb  = (bf16_t*)alloc((size_t)H * H * 2);
    bf16_t* Tt  = (bf16_t*)alloc((size_t)H * H * 2);
    bf16_t* Sva = (bf16_t*)alloc((size_t)H * H * 2);
    bf16_t* Svb = (bf16_t*)alloc((size_t)H * H * 2);
    bf16_t* HsA = (bf16_t*)alloc((size_t)NC * B * H * 2);
    bf16_t* HsB = (bf16_t*)alloc((size_t)NC * B * H * 2);
    bf16_t* hA  = (bf16_t*)alloc((size_t)16 * H * 2);
    bf16_t* hB  = (bf16_t*)alloc((size_t)16 * H * 2);
    bf16_t* Gam = (bf16_t*)alloc((size_t)NC * B * H * 2);

    auto direct = [](long long ld, long long base) -> RowMap {
        return RowMap{base, 8 * ld, ld, 3};   // row i -> base + i*ld
    };
    const RowMap noMap{0, 0, 0, 3};

    auto gemm = [&](const void* A, RowMap mA, const bf16_t* Bw_,
                    const bf16_t* Add, RowMap mC, void* D, RowMap mD,
                    int M, int N, int K, bool aF32, bool dF32) {
        int tiles  = ((M + 15) >> 4) * (N >> 6);
        int blocks = (tiles + 7) >> 3;
        if (aF32)
            gemm_wmma_bf16<true,  false><<<blocks, 256, 0, stream>>>(A, mA, Bw_, Add, mC, D, mD, M, N, K);
        else if (dF32)
            gemm_wmma_bf16<false, true ><<<blocks, 256, 0, stream>>>(A, mA, Bw_, Add, mC, D, mD, M, N, K);
        else
            gemm_wmma_bf16<false, false><<<blocks, 256, 0, stream>>>(A, mA, Bw_, Add, mC, D, mD, M, N, K);
    };

    // ---- weight fp32 -> bf16 ----
    f32_to_bf16_k<<<(H * DIN + 255) / 256, 256, 0, stream>>>(U, Ub, H * DIN);
    f32_to_bf16_k<<<(H * H + 255) / 256, 256, 0, stream>>>(W, Wb, H * H);
    f32_to_bf16_k<<<(H * H + 255) / 256, 256, 0, stream>>>(V, Vb, H * H);
    f32_to_bf16_k<<<(H * H + 255) / 256, 256, 0, stream>>>(Mw, Mb, H * H);
    f32_to_bf16_k<<<(DOUT * H + 255) / 256, 256, 0, stream>>>(cT, cTb, DOUT * H);

    // ---- xu = x @ U^T ; x rows i=b*L+t -> xu row t*8+b ----
    gemm(x, direct(DIN, 0), Ub, nullptr, noMap, xu,
         RowMap{0, (long long)H, 8LL * H, 11}, B * L, H, DIN, true, false);

    // ---- W^128 via 7 squarings (S <- S*S using T = S^T, D = S @ T^T) ----
    const bf16_t* Scur = Wb;
    { bf16_t* pp[2] = {Sa, Sb};
      for (int it = 0; it < 7; ++it) {
          transpose_bf16<<<dim3(H / 32, H / 32), 256, 0, stream>>>(Scur, Tt, H, H);
          gemm(Scur, direct(H, 0), Tt, nullptr, noMap, pp[it & 1], direct(H, 0), H, H, H, false, false);
          Scur = pp[it & 1];
      } }
    const bf16_t* W128 = Scur;

    // ---- encoder chunk-local scan: 128 batched steps, M = 16 chunks * 8 = 128 ----
    zero_bf16_k<<<(NC * B * H + 255) / 256, 256, 0, stream>>>(HsA, NC * B * H);
    { bf16_t* hs[2] = {HsA, HsB};
      for (int s = 0; s < CENC; ++s) {
          RowMap mC{(long long)s * 8 * H, (long long)CENC * 8 * H, (long long)H, 3};
          gemm(hs[s & 1], direct(H, 0), Wb, xu, mC, hs[(s & 1) ^ 1], direct(H, 0),
               NC * B, H, H, false, false);
      } }  // 128 steps end in HsA

    // ---- combine: carry_j = r_j + carry_{j-1} @ (W^128)^T ----
    hipMemcpyAsync(hA, HsA, (size_t)B * H * 2, hipMemcpyDeviceToDevice, stream); // carry = r_0
    { bf16_t* hv[2] = {hA, hB};
      for (int j = 1; j < NC; ++j) {
          RowMap mC{(long long)j * 8 * H, 8LL * H, (long long)H, 3};
          gemm(hv[(j - 1) & 1], direct(H, 0), W128, HsA, mC, hv[j & 1], direct(H, 0),
               B, H, H, false, false);
      } }
    const bf16_t* hEnc = hB;   // j=15 wrote hB

    // ---- V^32 via 5 squarings ----
    Scur = Vb;
    { bf16_t* pp[2] = {Sva, Svb};
      for (int it = 0; it < 5; ++it) {
          transpose_bf16<<<dim3(H / 32, H / 32), 256, 0, stream>>>(Scur, Tt, H, H);
          gemm(Scur, direct(H, 0), Tt, nullptr, noMap, pp[it & 1], direct(H, 0), H, H, H, false, false);
          Scur = pp[it & 1];
      } }
    const bf16_t* V32 = Scur;

    // ---- decoder seeds: Gam rows j*8+b = g0 @ ((V^32)^T)^j ----
    gemm(hEnc, direct(H, 0), Mb, nullptr, noMap, Gam, direct(H, 0), B, H, H, false, false);
    for (int j = 1; j < NC; ++j)
        gemm(Gam + (size_t)(j - 1) * 8 * H, direct(H, 0), V32, nullptr, noMap,
             Gam + (size_t)j * 8 * H, direct(H, 0), B, H, H, false, false);

    // ---- decoder batched steps: 32 steps, M=128, writing outs rows t*8+b ----
    for (int s = 1; s <= CDEC; ++s) {
        RowMap mD{(long long)(s - 1) * 8 * H, (long long)CDEC * 8 * H, (long long)H, 3};
        if (s == 1)
            gemm(Gam, direct(H, 0), Vb, nullptr, noMap, outs, mD, NC * B, H, H, false, false);
        else {
            RowMap mA{(long long)(s - 2) * 8 * H, (long long)CDEC * 8 * H, (long long)H, 3};
            gemm(outs, mA, Vb, nullptr, noMap, outs, mD, NC * B, H, H, false, false);
        }
    }

    // ---- readout: out[b][t][o] = outs[t*8+b] @ cT^T (fp32 out) ----
    gemm(outs, direct(H, 0), cTb, nullptr, noMap, out,
         RowMap{0, (long long)DOUT, (long long)OUT_LEN * DOUT, 3},
         OUT_LEN * B, DOUT, H, false, true);
}
